// NNCL_76647986364457
// MI455X (gfx1250) — matrix operand order repaired and checked by reading, not compiled, not run on hardware
//
#include <hip/hip_runtime.h>
#include <hip/hip_bf16.h>

#define DIMK   128
#define RQ     131072
#define BROWS  1024
#define NNB    10
#define NCHUNK 16
#define CHUNKC (RQ / NCHUNK)     // 8192 columns per stage-1 chunk
#define MROWS  128               // rows per stage-1/2 block (8 waves x 16)
#define NT1    32                // stage-1 columns per iteration (2 N-tiles)
#define NCOLS2 10240             // B * NNB
#define QELEM  ((size_t)DIMK * (size_t)RQ)   // elements per queue

typedef __attribute__((ext_vector_type(16))) __bf16 v16bf;
typedef __attribute__((ext_vector_type(8)))  float  v8f;

union BFrag { unsigned int u[8]; v16bf v; };

__device__ __forceinline__ unsigned short f2bf(float f) {
  unsigned int u = __float_as_uint(f);
  u += 0x7FFFu + ((u >> 16) & 1u);          // round-to-nearest-even
  return (unsigned short)(u >> 16);
}
__device__ __forceinline__ unsigned int packbf(float lo, float hi) {
  return (unsigned int)f2bf(lo) | ((unsigned int)f2bf(hi) << 16);
}

// A-matrix fragment (16x32 bf16 per k-block, 4 k-blocks = K128), contiguous K order (B uses the same):
// lanes 0-15: M=lane, VGPR j0..7 = K {0,1..14,15}+kh*16 (contiguous)
__device__ __forceinline__ void loadA(BFrag Af[4], const unsigned short* fbf,
                                      int side, int rowBase, int lane) {
  int m = lane & 15, kh = lane >> 4;
  const uint4* arow = (const uint4*)(fbf + ((size_t)(side * BROWS + rowBase + m)) * DIMK);
  #pragma unroll
  for (int b = 0; b < 4; ++b) {
    uint4 lo = arow[b * 4 + kh * 2];      // bf16 K = b*32 + kh*16 .. +7
    uint4 hi = arow[b * 4 + kh * 2 + 1];  // bf16 K = b*32 + kh*16 + 8 .. +15
    Af[b].u[0] = lo.x; Af[b].u[1] = lo.y; Af[b].u[2] = lo.z; Af[b].u[3] = lo.w;
    Af[b].u[4] = hi.x; Af[b].u[5] = hi.y; Af[b].u[6] = hi.z; Af[b].u[7] = hi.w;
  }
}

__device__ __forceinline__ void loadB_swz(BFrag Bf[4], const uint4* Qv,
                                          int col, int kh) {
  #pragma unroll
  for (int b = 0; b < 4; ++b) {
    const uint4* p = Qv + ((size_t)(b * 2 + kh) * RQ + col) * 2;
    uint4 lo = p[0], hi = p[1];
    Bf[b].u[0] = lo.x; Bf[b].u[1] = lo.y; Bf[b].u[2] = lo.z; Bf[b].u[3] = lo.w;
    Bf[b].u[4] = hi.x; Bf[b].u[5] = hi.y; Bf[b].u[6] = hi.z; Bf[b].u[7] = hi.w;
  }
}

// ---------------- Kernel 0: queue f32 -> swizzled bf16 image ----------------
// Layout: for side s, K-group g (16 K), column col: 8 packed u32 K-pairs
// stored contiguously at u32 index  s*(QELEM/2) + (g*RQ + col)*8 + j,
// pair j holding bf16 for K = 16g+2j, 16g+2j+1.
__global__ void k_cvtq(const float* qX, const float* qY, unsigned int* q16) {
  int s = blockIdx.z, g = blockIdx.y;
  size_t col = (size_t)blockIdx.x * blockDim.x + threadIdx.x;
  const float* Q = (s == 0) ? qX : qY;
  unsigned int o[8];
  #pragma unroll
  for (int j = 0; j < 8; ++j) {
    float a = Q[(size_t)(16 * g + 2 * j) * RQ + col];
    float b = Q[(size_t)(16 * g + 2 * j + 1) * RQ + col];
    o[j] = packbf(a, b);
  }
  uint4* dst = (uint4*)(q16 + (size_t)s * (QELEM / 2)) + ((size_t)g * RQ + col) * 2;
  uint4 v0; v0.x = o[0]; v0.y = o[1]; v0.z = o[2]; v0.w = o[3];
  uint4 v1; v1.x = o[4]; v1.y = o[5]; v1.z = o[6]; v1.w = o[7];
  dst[0] = v0; dst[1] = v1;
}

// ---------------- Kernel 1: row L2-normalize + bf16 copy ----------------
__global__ void k_normalize(const float* fx, const float* fy,
                            float* fXn, unsigned short* fbf) {
  int side = blockIdx.x >> 10, row = blockIdx.x & 1023;
  const float* src = (side == 0 ? fx : fy) + (size_t)row * DIMK;
  __shared__ float red[DIMK];
  float v = src[threadIdx.x];
  red[threadIdx.x] = v * v;
  __syncthreads();
  for (int s = 64; s > 0; s >>= 1) {
    if ((int)threadIdx.x < s) red[threadIdx.x] += red[threadIdx.x + s];
    __syncthreads();
  }
  float nv = v * rsqrtf(red[0]);
  size_t o = ((size_t)(side * BROWS + row)) * DIMK + threadIdx.x;
  fXn[o] = nv;
  fbf[o] = f2bf(nv);
}

// ------- Kernel 2: fused  sim = fX @ queue  (WMMA bf16) + per-chunk top-10 -------
// grid (8 row-blocks, 16 chunks, 2 sides), 256 threads (8 waves x 16 rows each).
// 32 columns per iteration, double-buffered sim tiles, one barrier per iteration.
__global__ void k_sim_topk(const unsigned short* fbf, const unsigned int* q16,
                           float* candV, int* candI) {
  const int side  = blockIdx.z;
  const int chunk = blockIdx.y;
  const int rowBase = blockIdx.x * MROWS;
  // loss_X pairs feat_X with queue_Y: side 0 -> q16 region 1
  const uint4* Qv = (const uint4*)q16 + (size_t)(side ^ 1) * (QELEM / 8);

  const int lane = threadIdx.x & 31;
  const int w    = threadIdx.x >> 5;
  const int kh   = lane >> 4;

  BFrag Af[4];
  loadA(Af, fbf, side, rowBase + w * 16, lane);

  __shared__ float sims[2][MROWS * NT1];

  // register-resident top-10 (ascending, tv[0] = current min)
  const int srow = threadIdx.x >> 1;           // scan row 0..127
  const int shalf = threadIdx.x & 1;           // which 16 columns of the tile
  float tv[NNB]; int ti[NNB];
  #pragma unroll
  for (int j = 0; j < NNB; ++j) { tv[j] = -3.0e38f; ti[j] = 0; }

  const int chunkBase = chunk * CHUNKC;
  for (int it = 0; it < CHUNKC / NT1; ++it) {
    const int p = it & 1;
    const int colBase = chunkBase + it * NT1;
    const int col0 = colBase + (lane & 15);

    BFrag Bf0[4], Bf1[4];
    loadB_swz(Bf0, Qv, col0, kh);
    loadB_swz(Bf1, Qv, col0 + 16, kh);

    v8f c0 = {}, c1 = {};
    #pragma unroll
    for (int b = 0; b < 4; ++b) {
      c0 = __builtin_amdgcn_wmma_f32_16x16x32_bf16(false, Af[b].v, false, Bf0[b].v,
                                                   (short)0, c0, false, false);
      c1 = __builtin_amdgcn_wmma_f32_16x16x32_bf16(false, Af[b].v, false, Bf1[b].v,
                                                   (short)0, c1, false, false);
    }
    // D layout: VGPR i -> row i + 8*kh, column = lane&15
    #pragma unroll
    for (int i = 0; i < 8; ++i) {
      int rl = (w * 16 + i + 8 * kh) * NT1 + (lane & 15);
      sims[p][rl]      = c0[i];
      sims[p][rl + 16] = c1[i];
    }
    __syncthreads();

    // scan: 2 threads per row, 16 columns each, via 4 x ds_load_b128
    const float4* s4 = (const float4*)&sims[p][srow * NT1 + shalf * 16];
    #pragma unroll
    for (int q = 0; q < 4; ++q) {
      float4 vv = s4[q];
      #pragma unroll
      for (int e = 0; e < 4; ++e) {
        float v = (e == 0) ? vv.x : (e == 1) ? vv.y : (e == 2) ? vv.z : vv.w;
        if (v > tv[0]) {
          int gcol = colBase + shalf * 16 + q * 4 + e;
          #pragma unroll
          for (int j = 0; j < NNB; ++j) {
            if (j == NNB - 1 || tv[j + 1] >= v) { tv[j] = v; ti[j] = gcol; break; }
            tv[j] = tv[j + 1]; ti[j] = ti[j + 1];
          }
        }
      }
    }
  }

  size_t cb = ((((size_t)side * NCHUNK + chunk) * BROWS + (rowBase + srow)) * 2 + shalf) * NNB;
  #pragma unroll
  for (int j = 0; j < NNB; ++j) { candV[cb + j] = tv[j]; candI[cb + j] = ti[j]; }
}

// -------- Kernel 3: merge 16 chunks x 20 candidates -> final top-10; zero LSE accum --------
__global__ void k_merge_topk(const float* candV, const int* candI,
                             int* finalIdx, float* rowSE) {
  int t = blockIdx.x * blockDim.x + threadIdx.x;    // 0..2047
  int side = t >> 10, row = t & 1023;
  rowSE[t] = 0.0f;
  float prevV = 3.0e38f; int prevI = -1;
  size_t rb = ((size_t)side * NCHUNK * BROWS + row) * 20;  // stride between chunks = 1024*20
  for (int p = 0; p < NNB; ++p) {
    float bestV = -3.0e38f; int bestI = 0x7fffffff;
    for (int c = 0; c < NCHUNK; ++c) {
      size_t base = rb + (size_t)c * BROWS * 20;
      for (int e = 0; e < 20; ++e) {
        float v = candV[base + e]; int id = candI[base + e];
        bool lt_prev = (v < prevV) || (v == prevV && id > prevI);
        bool gt_best = (v > bestV) || (v == bestV && id < bestI);
        if (lt_prev && gt_best) { bestV = v; bestI = id; }
      }
    }
    finalIdx[(size_t)t * NNB + p] = bestI;
    prevV = bestV; prevI = bestI;
  }
}

// -------- Kernel 4: gather neighbor columns -> bf16, K-contiguous rows --------
__global__ void k_gather(const int* finalIdx, const float* qX, const float* qY,
                         unsigned short* nnb) {
  int j = blockIdx.x, side = blockIdx.y, k = threadIdx.x;
  const float* Q = (side == 0) ? qY : qX;
  int col = finalIdx[(size_t)side * NCOLS2 + j];
  nnb[((size_t)side * NCOLS2 + j) * DIMK + k] = f2bf(Q[(size_t)k * RQ + col]);
}

// -------- Kernel 5: sim2 = fX @ nn^T via WMMA; accumulate sum(exp(10*sim)) per row --------
// grid (8 row-blocks, 8 col-chunks of 1280, 2 sides), 256 threads
__global__ void k_lse(const unsigned short* fbf, const unsigned short* nnb, float* rowSE) {
  const int side = blockIdx.z;
  const int rowBase = blockIdx.x * MROWS;
  const int lane = threadIdx.x & 31;
  const int w    = threadIdx.x >> 5;
  const int kh   = lane >> 4;

  BFrag Af[4];
  loadA(Af, fbf, side, rowBase + w * 16, lane);

  float acc[8];
  #pragma unroll
  for (int i = 0; i < 8; ++i) acc[i] = 0.0f;

  for (int it = 0; it < 80; ++it) {
    const int col = blockIdx.y * 1280 + it * 16 + (lane & 15);
    const uint4* nrow = (const uint4*)(nnb) + ((size_t)side * NCOLS2 + col) * 16;
    BFrag Bf[4];
    #pragma unroll
    for (int b = 0; b < 4; ++b) {
      uint4 lo = nrow[b * 4 + kh * 2];
      uint4 hi = nrow[b * 4 + kh * 2 + 1];
      Bf[b].u[0] = lo.x; Bf[b].u[1] = lo.y; Bf[b].u[2] = lo.z; Bf[b].u[3] = lo.w;
      Bf[b].u[4] = hi.x; Bf[b].u[5] = hi.y; Bf[b].u[6] = hi.z; Bf[b].u[7] = hi.w;
    }
    v8f c = {};
    #pragma unroll
    for (int b = 0; b < 4; ++b)
      c = __builtin_amdgcn_wmma_f32_16x16x32_bf16(false, Af[b].v, false, Bf[b].v,
                                                  (short)0, c, false, false);
    #pragma unroll
    for (int i = 0; i < 8; ++i) acc[i] += __expf(c[i] * 10.0f);
  }

  __shared__ float lse[MROWS];
  if (threadIdx.x < MROWS) lse[threadIdx.x] = 0.0f;
  __syncthreads();
  #pragma unroll
  for (int i = 0; i < 8; ++i)
    atomicAdd(&lse[w * 16 + i + 8 * kh], acc[i]);
  __syncthreads();
  if (threadIdx.x < MROWS)
    atomicAdd(&rowSE[(size_t)side * BROWS + rowBase + threadIdx.x], lse[threadIdx.x]);
}

// -------- Kernel 6: per-row loss = sum_j(LSE - pos_j) --------
__global__ void k_rowloss(const float* fXn, const int* finalIdx, const float* rowSE,
                          const float* qX, const float* qY, float* rowLoss) {
  int t = blockIdx.x * blockDim.x + threadIdx.x;    // 0..2047
  int side = t >> 10, row = t & 1023;
  const float* f = fXn + ((size_t)side * BROWS + row) * DIMK;
  const float* Q = (side == 0) ? qY : qX;
  float s = 0.0f;
  for (int j = 0; j < NNB; ++j) {
    int col = finalIdx[(size_t)t * NNB + j];
    float d = 0.0f;
    for (int k = 0; k < DIMK; ++k) d += f[k] * Q[(size_t)k * RQ + col];
    s += d * 10.0f;
  }
  rowLoss[t] = (float)NNB * logf(rowSE[t]) - s;
}

// -------- Kernel 7: final reduce -> loss scalar --------
__global__ void k_loss_reduce(const float* rowLoss, float* out) {
  __shared__ float red[256];
  float s = 0.0f;
  for (int i = threadIdx.x; i < 2 * BROWS; i += 256) s += rowLoss[i];
  red[threadIdx.x] = s;
  __syncthreads();
  for (int st = 128; st > 0; st >>= 1) {
    if ((int)threadIdx.x < st) red[threadIdx.x] += red[threadIdx.x + st];
    __syncthreads();
  }
  if (threadIdx.x == 0) out[0] = red[0] / (float)BROWS;
}

// -------- Kernel 8: copy queue columns [1024, RQ) to output --------
__global__ void k_copy_queue(const float* qX, const float* qY, float* out) {
  int side = blockIdx.z, d = blockIdx.y;
  const float* Q = (side == 0) ? qX : qY;          // new_queue_X derives from queue_X
  size_t r = 1024 + ((size_t)blockIdx.x * blockDim.x + threadIdx.x) * 4;
  const float* src = Q + (size_t)d * RQ + r;
  float* dst = out + 1 + (size_t)side * QELEM + (size_t)d * RQ + r;
  dst[0] = src[0]; dst[1] = src[1]; dst[2] = src[2]; dst[3] = src[3];
}

// -------- Kernel 9: first 1024 columns <- normalized features (transposed) --------
__global__ void k_head(const float* fXn, float* out) {
  int g = blockIdx.x * blockDim.x + threadIdx.x;   // 0 .. 2*128*1024-1
  int side = g >> 17;
  int rem = g & 131071;
  int d = rem >> 10, r = rem & 1023;
  out[1 + (size_t)side * QELEM + (size_t)d * RQ + r] =
      fXn[((size_t)side * BROWS + r) * DIMK + d];
}

extern "C" void kernel_launch(void* const* d_in, const int* in_sizes, int n_in,
                              void* d_out, int out_size, void* d_ws, size_t ws_size,
                              hipStream_t stream) {
  const float* fx = (const float*)d_in[0];
  const float* fy = (const float*)d_in[1];
  const float* qX = (const float*)d_in[2];
  const float* qY = (const float*)d_in[3];
  float* out = (float*)d_out;

  char* ws = (char*)d_ws;
  float*          fXn      = (float*)(ws + 0);                 // 2*1024*128 f32   (1.00 MB)
  unsigned short* fbf      = (unsigned short*)(ws + 1048576);  // same, bf16       (0.50 MB)
  float*          candV    = (float*)(ws + 1572864);           // 2*16*1024*20 f32 (2.50 MB)
  int*            candI    = (int*)  (ws + 4194304);           //                  (2.50 MB)
  int*            finalIdx = (int*)  (ws + 6815744);           // 2*1024*10        (80 KB)
  unsigned short* nnb      = (unsigned short*)(ws + 6897664);  // 2*10240*128 bf16 (5.00 MB)
  float*          rowSE    = (float*)(ws + 12140544);          // 2048 f32
  float*          rowLoss  = (float*)(ws + 12148736);          // 2048 f32
  unsigned int*   q16      = (unsigned int*)(ws + 12156928);   // 2 queues bf16 swizzled (64 MB)

  k_cvtq<<<dim3(RQ / 256, DIMK / 16, 2), 256, 0, stream>>>(qX, qY, q16);
  k_normalize<<<2048, 128, 0, stream>>>(fx, fy, fXn, fbf);
  k_sim_topk<<<dim3(BROWS / MROWS, NCHUNK, 2), 256, 0, stream>>>(fbf, q16, candV, candI);
  k_merge_topk<<<8, 256, 0, stream>>>(candV, candI, finalIdx, rowSE);
  k_gather<<<dim3(NCOLS2, 2), 128, 0, stream>>>(finalIdx, qX, qY, nnb);
  k_lse<<<dim3(BROWS / MROWS, 8, 2), 256, 0, stream>>>(fbf, nnb, rowSE);
  k_rowloss<<<8, 256, 0, stream>>>(fXn, finalIdx, rowSE, qX, qY, rowLoss);
  k_loss_reduce<<<1, 256, 0, stream>>>(rowLoss, out);
  k_copy_queue<<<dim3((RQ - 1024) / 4 / 256, DIMK, 2), 256, 0, stream>>>(qX, qY, out);
  k_head<<<2 * DIMK * BROWS / 256, 256, 0, stream>>>(fXn, out);
}
